// MolecularProdRuleEmbeddingUsingFeatures_5076651344549
// MI455X (gfx1250) — compile-verified, matrix operand order checked by reading, not run on hardware
//
#include <hip/hip_runtime.h>
#include <hip/hip_bf16.h>

typedef __bf16 bf16;
typedef __attribute__((ext_vector_type(16))) __bf16 v16bf;
typedef __attribute__((ext_vector_type(8)))  __bf16 v8bf;
typedef __attribute__((ext_vector_type(8)))  float  v8f;

#define RULES 200
#define NN    16
#define FF    512
#define DD    128
#define NLAYERS 3
#define BB    32
#define LL    128

// A lane's 16-half WMMA fragment (16-bit A 16x32 / B 32x16 layout, ISA 7.12.2)
// is two contiguous 16-byte chunks of a row-major [row][K] buffer:
//   halves 0..7  at K = g*8 .. g*8+7
//   halves 8..15 at K = 16+g*8 .. 16+g*8+7        (g = lane/16)
// base points at [row][g*8]; second chunk is 16 halves further.
__device__ __forceinline__ v16bf load_frag(const bf16* base) {
    v8bf lo = *(const v8bf*)(base);
    v8bf hi = *(const v8bf*)(base + 16);
    return __builtin_shufflevector(lo, hi,
        0, 1, 2, 3, 4, 5, 6, 7, 8, 9, 10, 11, 12, 13, 14, 15);
}

// ---------------------------------------------------------------------------
// Kernel 0: one-time f32 -> bf16 convert + transpose of the weights.
// WllT[l][n][k] = Wll[l][k][n]   (so B-fragments are contiguous along K)
// ---------------------------------------------------------------------------
__global__ void convert_weights_kernel(const float* __restrict__ Wll,
                                       const float* __restrict__ Wlo,
                                       bf16* __restrict__ WllT,
                                       bf16* __restrict__ WloT) {
    int i = blockIdx.x * blockDim.x + threadIdx.x;
    if (i < NLAYERS * FF * FF) {
        int l = i / (FF * FF), rem = i % (FF * FF);
        int k = rem / FF, n = rem % FF;
        WllT[(size_t)l * FF * FF + (size_t)n * FF + k] = (bf16)Wll[i];
    }
    if (i < NLAYERS * FF * DD) {
        int l = i / (FF * DD), rem = i % (FF * DD);
        int k = rem / DD, n = rem % DD;
        WloT[(size_t)l * DD * FF + (size_t)n * FF + k] = (bf16)Wlo[i];
    }
}

// ---------------------------------------------------------------------------
// Kernel 1: per-rule 3-layer GNN. One 256-thread block (8 wave32) per rule.
// ---------------------------------------------------------------------------
__global__ __launch_bounds__(256) void rule_gnn_kernel(
    const float* __restrict__ adj,        // [R,16,16]
    const float* __restrict__ init_embed, // [R,16,512]
    const float* __restrict__ node_mask,  // [R,16]
    const bf16*  __restrict__ WllT,       // [3,512(n),512(k)] transposed
    const float* __restrict__ b_ll,       // [3,512]
    const bf16*  __restrict__ WloT,       // [3,128(n),512(k)] transposed
    const float* __restrict__ b_lo,       // [3,128]
    float* __restrict__ out_rule)         // [R,128]
{
    __shared__ bf16  sA[NN * 32];      // adjacency, row-major [m][K], K pad 32
    __shared__ bf16  sET[FF * 32];     // embedding transposed [f][node], pad 32
    __shared__ bf16  sMsg[NN * FF];    // msg row-major [m][f]
    __shared__ float sMask[NN];

    const int r    = blockIdx.x;
    const int tid  = threadIdx.x;
    const int wave = tid >> 5;
    const int lane = tid & 31;
    const int g    = lane >> 4;   // half-wave
    const int ln   = lane & 15;   // row/col position within half-wave
    const int gk   = g * 8;       // K chunk offset for this half-wave

    // ---- stage adjacency (padded), mask, transposed embedding ----
    {
        int m = tid >> 4, k = tid & 15;        // 256 threads == 16x16
        sA[m * 32 + k]      = (bf16)adj[r * 256 + m * 16 + k];
        sA[m * 32 + 16 + k] = (bf16)0.0f;
    }
    if (tid < NN) sMask[tid] = node_mask[r * NN + tid];
    for (int i = tid; i < NN * FF; i += 256) {
        int m = i / FF, f = i % FF;
        sET[f * 32 + m] = (bf16)init_embed[(size_t)r * NN * FF + i];
    }
    for (int i = tid; i < FF * 16; i += 256) {   // zero node-pad rows 16..31
        int f = i >> 4, m = i & 15;
        sET[f * 32 + 16 + m] = (bf16)0.0f;
    }
    __syncthreads();

    // Wave owns output columns [wave*16, wave*16+16); lane ln is the column.
    float outacc = 0.0f;

    for (int l = 0; l < NLAYERS; ++l) {
        // ---------- Phase 1: msg = a @ e -----------------------------------
        // A = adjacency (16x32 padded), B tile = sET columns (zero-padded K).
        v16bf afrag = load_frag(&sA[ln * 32 + gk]);
        for (int t = wave; t < FF / 16; t += 8) {
            v16bf bfrag = load_frag(&sET[(t * 16 + ln) * 32 + gk]);
            v8f c = {};
            c = __builtin_amdgcn_wmma_f32_16x16x32_bf16(
                    false, afrag, false, bfrag, (short)0, c, false, false);
            #pragma unroll
            for (int i = 0; i < 8; ++i)        // D: VGPR i -> row g*8+i
                sMsg[(g * 8 + i) * FF + t * 16 + ln] = (bf16)c[i];
        }
        __syncthreads();

        // ---------- Phase 2+3 fused: one pass over msg A-fragments ---------
        //   acc0..3 : e' tiles of msg @ W_ll   (wave covers 4 of 32 N-tiles)
        //   acclo   : out tile  of msg @ W_lo  (wave covers 1 of 8 N-tiles)
        // Each A-fragment load feeds 5 WMMAs.
        {
            v8f acc0 = {}, acc1 = {}, acc2 = {}, acc3 = {}, acclo = {};
            const bf16* w0 =
                &WllT[((size_t)l * FF + wave * 16 + ln) * FF + gk];
            const bf16* w1 = w0 + (size_t)8  * 16 * FF;
            const bf16* w2 = w0 + (size_t)16 * 16 * FF;
            const bf16* w3 = w0 + (size_t)24 * 16 * FF;
            const bf16* wlo =
                &WloT[((size_t)l * DD + wave * 16 + ln) * FF + gk];

            for (int ks = 0; ks < FF / 32; ++ks) {
                v16bf af = load_frag(&sMsg[ln * FF + ks * 32 + gk]);
                if (ks + 1 < FF / 32) {        // pull next K panel near
                    __builtin_prefetch(w0  + (ks + 1) * 32, 0, 3);
                    __builtin_prefetch(wlo + (ks + 1) * 32, 0, 3);
                }
                v16bf b0 = load_frag(w0 + ks * 32);
                acc0 = __builtin_amdgcn_wmma_f32_16x16x32_bf16(
                        false, af, false, b0, (short)0, acc0, false, false);
                v16bf b1 = load_frag(w1 + ks * 32);
                acc1 = __builtin_amdgcn_wmma_f32_16x16x32_bf16(
                        false, af, false, b1, (short)0, acc1, false, false);
                v16bf b2 = load_frag(w2 + ks * 32);
                acc2 = __builtin_amdgcn_wmma_f32_16x16x32_bf16(
                        false, af, false, b2, (short)0, acc2, false, false);
                v16bf b3 = load_frag(w3 + ks * 32);
                acc3 = __builtin_amdgcn_wmma_f32_16x16x32_bf16(
                        false, af, false, b3, (short)0, acc3, false, false);
                v16bf bl = load_frag(wlo + ks * 32);
                acclo = __builtin_amdgcn_wmma_f32_16x16x32_bf16(
                        false, af, false, bl, (short)0, acclo, false, false);
            }

            // out epilogue: bias + tanh + row-mask + row reduction
            {
                float bias = b_lo[l * DD + wave * 16 + ln];
                float part = 0.0f;
                #pragma unroll
                for (int i = 0; i < 8; ++i)
                    part += tanhf(acclo[i] + bias) * sMask[g * 8 + i];
                outacc += part;
            }

            // e' epilogue: bias + relu, packed 16-byte LDS store per tile
            v8f accs[4] = {acc0, acc1, acc2, acc3};
            #pragma unroll
            for (int u = 0; u < 4; ++u) {
                int t = wave + u * 8;
                float bias = b_ll[l * FF + t * 16 + ln];
                v8bf pk;
                #pragma unroll
                for (int i = 0; i < 8; ++i) {
                    float v = accs[u][i] + bias;
                    pk[i] = (bf16)(v > 0.0f ? v : 0.0f);
                }
                // column f = t*16+ln, rows m = g*8 .. g*8+7 (contiguous)
                *(v8bf*)(&sET[(t * 16 + ln) * 32 + gk]) = pk;
            }
        }
        __syncthreads();   // sET / sMsg reuse barrier for next layer
    }

    // combine the two half-wave row-partials (rows 0-7 vs 8-15)
    float total = outacc + __shfl_xor(outacc, 16, 32);
    if (g == 0)
        out_rule[r * DD + wave * 16 + ln] = total;
}

// ---------------------------------------------------------------------------
// Kernel 2: per-token gather with padding-sentinel zeroing (float4 wide).
// ---------------------------------------------------------------------------
__global__ void scatter_out_kernel(const int* __restrict__ idx,
                                   const float4* __restrict__ out_rule,
                                   float4* __restrict__ out) {
    int token = blockIdx.x;          // B*L tokens
    int q     = threadIdx.x;         // D/4 = 32 float4s
    int r     = idx[token];
    float4 v  = (r < RULES) ? out_rule[r * (DD / 4) + q]
                            : make_float4(0.f, 0.f, 0.f, 0.f);
    out[(size_t)token * (DD / 4) + q] = v;
}

// ---------------------------------------------------------------------------
extern "C" void kernel_launch(void* const* d_in, const int* in_sizes, int n_in,
                              void* d_out, int out_size, void* d_ws, size_t ws_size,
                              hipStream_t stream) {
    const int*   idx        = (const int*)  d_in[0];
    const float* adj        = (const float*)d_in[1];
    const float* init_embed = (const float*)d_in[2];
    const float* node_mask  = (const float*)d_in[3];
    const float* W_ll       = (const float*)d_in[4];
    const float* b_ll       = (const float*)d_in[5];
    const float* W_lo       = (const float*)d_in[6];
    const float* b_lo       = (const float*)d_in[7];
    float* out = (float*)d_out;

    // Workspace layout: [WllT_bf | WloT_bf | out_rule]
    char* ws = (char*)d_ws;
    bf16*  WllT     = (bf16*)ws;
    size_t off      = (size_t)NLAYERS * FF * FF * sizeof(bf16);
    bf16*  WloT     = (bf16*)(ws + off);
    off            += (size_t)NLAYERS * FF * DD * sizeof(bf16);
    float* out_rule = (float*)(ws + off);

    const int nconv = NLAYERS * FF * FF;   // covers Wlo range too
    convert_weights_kernel<<<(nconv + 255) / 256, 256, 0, stream>>>(
        W_ll, W_lo, WllT, WloT);

    rule_gnn_kernel<<<RULES, 256, 0, stream>>>(
        adj, init_embed, node_mask, WllT, b_ll, WloT, b_lo, out_rule);

    scatter_out_kernel<<<BB * LL, DD / 4, 0, stream>>>(
        idx, (const float4*)out_rule, (float4*)out);
}
